// HybridConvKernelNet_65481071410365
// MI455X (gfx1250) — compile-verified, matrix operand order checked by reading, not run on hardware
//
#include <hip/hip_runtime.h>
#include <hip/hip_bf16.h>
#include <math.h>

typedef __attribute__((ext_vector_type(16))) _Float16 v16h;
typedef __attribute__((ext_vector_type(8)))  _Float16 v8h;
typedef __attribute__((ext_vector_type(8)))  float    v8f;

#define BATCH   128
#define FC1_K   55815                        // 15*61*61
#define KPAD    57344                        // 64 splits * 28 steps * K32, aligned
#define FC1_N   120
#define NPAD    128                          // padded N (extra W rows zeroed)
#define KSPLIT  64
#define KSTEPS  (KPAD / 32)                  // 1792
#define STEPS_PER (KSTEPS / KSPLIT)          // 28 (even, exact for every split)

// ---------------- conv1 (3->6, 5x5, s2, pad1) + relu + maxpool2x2 s1 --------
__global__ void conv1_pool_relu(const float* __restrict__ x,
                                const float* __restrict__ w,
                                const float* __restrict__ bias,
                                float* __restrict__ out) {
    __shared__ float sw[6 * 3 * 5 * 5 + 6];
    for (int i = threadIdx.x; i < 450; i += blockDim.x) sw[i] = w[i];
    for (int i = threadIdx.x; i < 6;   i += blockDim.x) sw[450 + i] = bias[i];
    __syncthreads();

    int t = blockIdx.x * blockDim.x + threadIdx.x;
    const int total = BATCH * 6 * 123 * 123;
    if (t >= total) return;
    int ow = t % 123; int tmp = t / 123;
    int oh = tmp % 123; tmp /= 123;
    int oc = tmp % 6;  int b = tmp / 6;

    const float* xb = x + (long long)b * 3 * 250 * 250;
    const float* wk = sw + oc * 75;
    float mv = -INFINITY;
    #pragma unroll
    for (int ph = 0; ph < 2; ++ph) {
        #pragma unroll
        for (int pw = 0; pw < 2; ++pw) {
            int ch = oh + ph, cw = ow + pw;
            float acc = sw[450 + oc];
            for (int ic = 0; ic < 3; ++ic) {
                const float* xc = xb + ic * 250 * 250;
                const float* wc = wk + ic * 25;
                #pragma unroll
                for (int kh = 0; kh < 5; ++kh) {
                    int ih = ch * 2 - 1 + kh;
                    if (ih < 0 || ih >= 250) continue;
                    #pragma unroll
                    for (int kw = 0; kw < 5; ++kw) {
                        int iw = cw * 2 - 1 + kw;
                        if (iw < 0 || iw >= 250) continue;
                        acc = fmaf(xc[ih * 250 + iw], wc[kh * 5 + kw], acc);
                    }
                }
            }
            mv = fmaxf(mv, acc);
        }
    }
    out[t] = fmaxf(mv, 0.0f);
}

// ---------------- conv2 (6->15, 3x3, s2, pad1) + relu + pool -> f16 A -------
__global__ void conv2_pool_relu(const float* __restrict__ in,
                                const float* __restrict__ w,
                                const float* __restrict__ bias,
                                _Float16* __restrict__ A16) {
    __shared__ float sw[15 * 6 * 9 + 15];
    for (int i = threadIdx.x; i < 810; i += blockDim.x) sw[i] = w[i];
    for (int i = threadIdx.x; i < 15;  i += blockDim.x) sw[810 + i] = bias[i];
    __syncthreads();

    int t = blockIdx.x * blockDim.x + threadIdx.x;
    const int total = BATCH * 15 * 61 * 61;
    if (t >= total) return;
    int ow = t % 61; int tmp = t / 61;
    int oh = tmp % 61; tmp /= 61;
    int oc = tmp % 15; int b = tmp / 15;

    const float* xb = in + (long long)b * 6 * 123 * 123;
    float mv = -INFINITY;
    #pragma unroll
    for (int ph = 0; ph < 2; ++ph) {
        #pragma unroll
        for (int pw = 0; pw < 2; ++pw) {
            int ch = oh + ph, cw = ow + pw;
            float acc = sw[810 + oc];
            for (int ic = 0; ic < 6; ++ic) {
                const float* xc = xb + ic * 123 * 123;
                const float* wc = sw + oc * 54 + ic * 9;
                #pragma unroll
                for (int kh = 0; kh < 3; ++kh) {
                    int ih = ch * 2 - 1 + kh;
                    if (ih < 0 || ih >= 123) continue;
                    #pragma unroll
                    for (int kw = 0; kw < 3; ++kw) {
                        int iw = cw * 2 - 1 + kw;
                        if (iw < 0 || iw >= 123) continue;
                        acc = fmaf(xc[ih * 123 + iw], wc[kh * 3 + kw], acc);
                    }
                }
            }
            mv = fmaxf(mv, acc);
        }
    }
    int inner = (oc * 61 + oh) * 61 + ow;              // NCHW flatten index
    A16[(long long)b * KPAD + inner] = (_Float16)fmaxf(mv, 0.0f);
}

// Zero the K-pad tail of each A row ([FC1_K, KPAD)).
__global__ void pad_a16(_Float16* __restrict__ A16) {
    int i = blockIdx.x * blockDim.x + threadIdx.x;
    const int padw = KPAD - FC1_K;                     // 1529
    if (i >= BATCH * padw) return;
    int row = i / padw, k = FC1_K + i % padw;
    A16[(long long)row * KPAD + k] = (_Float16)0.f;
}

// Convert fc1_w (120 x FC1_K f32) into padded 128 x KPAD f16 (zero-filled).
__global__ void prep_w16(const float* __restrict__ W,
                         _Float16* __restrict__ W16) {
    long long i = (long long)blockIdx.x * blockDim.x + threadIdx.x;
    if (i >= (long long)NPAD * KPAD) return;
    int n = (int)(i / KPAD), k = (int)(i % KPAD);
    float v = (n < FC1_N && k < FC1_K) ? W[(long long)n * FC1_K + k] : 0.f;
    W16[i] = (_Float16)v;
}

// ---------------- fc1: C(128,120) = A(128,K) * W(120,K)^T via WMMA ----------
// One wave per (16x16 tile, K-split); 64 splits * 64 tiles = 4096 waves.
// Guard-free (padded/aligned f16) and manually double-buffered: two disjoint
// fragment register sets, each WMMA consumes fragments loaded one half-body
// earlier while 4 newer b128 loads stay in flight (wait loadcnt<=4, not 0).
// Fragment layout (ISA 7.12.2, wave32, 16-bit A 16x32): lane L holds row
// M = L&15, K base = (L>=16 ? 8 : 0); element e -> K = base + e + (e&8).
// B loaded symmetrically per-lane from rows of W (columns of B = W^T).
__global__ void fc1_wmma(const _Float16* __restrict__ A16,
                         const _Float16* __restrict__ W16,
                         float* __restrict__ Cpart) {
    int blk  = blockIdx.x;
    int mt   = blk & 7;
    int nt   = (blk >> 3) & 7;
    int ks   = blk >> 6;                     // 0..63
    int lane = threadIdx.x;
    int half = lane >> 4;
    int l15  = lane & 15;

    const _Float16* ap = A16 + (long long)(mt * 16 + l15) * KPAD
                             + (long long)(ks * STEPS_PER) * 32 + half * 8;
    const _Float16* wp = W16 + (long long)(nt * 16 + l15) * KPAD
                             + (long long)(ks * STEPS_PER) * 32 + half * 8;

    v8f c = {0.f, 0.f, 0.f, 0.f, 0.f, 0.f, 0.f, 0.f};

    // Set 0: fragments for step 0.
    v8h A0 = *(const v8h*)(ap);
    v8h A1 = *(const v8h*)(ap + 16);
    v8h B0 = *(const v8h*)(wp);
    v8h B1 = *(const v8h*)(wp + 16);

    for (int it = 0; it < STEPS_PER; it += 2) {
        // Issue set-1 loads (step it+1) before consuming set 0.
        int off1 = (it + 1) * 32;
        v8h C0 = *(const v8h*)(ap + off1);
        v8h C1 = *(const v8h*)(ap + off1 + 16);
        v8h D0 = *(const v8h*)(wp + off1);
        v8h D1 = *(const v8h*)(wp + off1 + 16);

        v16h a0v = __builtin_shufflevector(A0, A1, 0, 1, 2, 3, 4, 5, 6, 7,
                                                   8, 9, 10, 11, 12, 13, 14, 15);
        v16h b0v = __builtin_shufflevector(B0, B1, 0, 1, 2, 3, 4, 5, 6, 7,
                                                   8, 9, 10, 11, 12, 13, 14, 15);
        c = __builtin_amdgcn_wmma_f32_16x16x32_f16(
                false, a0v, false, b0v, (short)0, c, false, false);

        // Issue set-0 loads for step it+2 (last iteration: harmless reload of
        // step 0; data discarded after the loop). Uniform, branch-free.
        int off2 = (it + 2 < STEPS_PER) ? (it + 2) * 32 : 0;
        A0 = *(const v8h*)(ap + off2);
        A1 = *(const v8h*)(ap + off2 + 16);
        B0 = *(const v8h*)(wp + off2);
        B1 = *(const v8h*)(wp + off2 + 16);

        v16h a1v = __builtin_shufflevector(C0, C1, 0, 1, 2, 3, 4, 5, 6, 7,
                                                   8, 9, 10, 11, 12, 13, 14, 15);
        v16h b1v = __builtin_shufflevector(D0, D1, 0, 1, 2, 3, 4, 5, 6, 7,
                                                   8, 9, 10, 11, 12, 13, 14, 15);
        c = __builtin_amdgcn_wmma_f32_16x16x32_f16(
                false, a1v, false, b1v, (short)0, c, false, false);
    }

    float* dst = Cpart + (long long)ks * (BATCH * NPAD);
    #pragma unroll
    for (int r = 0; r < 8; ++r) {
        int row = mt * 16 + r + half * 8;    // C layout per ISA (32-bit 16x16)
        int col = nt * 16 + l15;
        dst[row * NPAD + col] = c[r];
    }
}

// ---------------- deterministic split-K reduction ---------------------------
__global__ void fc1_reduce(const float* __restrict__ Cpart,
                           float* __restrict__ Cfin) {
    int i = blockIdx.x * blockDim.x + threadIdx.x;
    if (i >= BATCH * NPAD) return;
    float s = 0.f;
    for (int ks = 0; ks < KSPLIT; ++ks)
        s += Cpart[(long long)ks * (BATCH * NPAD) + i];
    Cfin[i] = s;
}

// ---------------- epilogue: bias+relu, fc2 rows 0..3, cos-kernel head -------
__global__ void head_kernel(const float* __restrict__ Cfin,
                            const float* __restrict__ fc1_b,
                            const float* __restrict__ fc2_w,
                            const float* __restrict__ fc2_b,
                            const float* __restrict__ support,
                            const float* __restrict__ proj_w,
                            const float* __restrict__ proj_b,
                            float* __restrict__ out) {
    int row = blockIdx.x * blockDim.x + threadIdx.x;
    if (row >= BATCH) return;
    const float* crow = Cfin + row * NPAD;
    float q0 = fc2_b[0], q1 = fc2_b[1], q2 = fc2_b[2], q3 = fc2_b[3];
    for (int j = 0; j < FC1_N; ++j) {
        float v = fmaxf(crow[j] + fc1_b[j], 0.f);
        q0 = fmaf(fc2_w[0 * FC1_N + j], v, q0);
        q1 = fmaf(fc2_w[1 * FC1_N + j], v, q1);
        q2 = fmaf(fc2_w[2 * FC1_N + j], v, q2);
        q3 = fmaf(fc2_w[3 * FC1_N + j], v, q3);
    }
    float logit = proj_b[0];
    for (int s = 0; s < 10; ++s) {
        float p = cosf(0.5f * (q0 - support[s * 4 + 0]))
                * cosf(0.5f * (q1 - support[s * 4 + 1]))
                * cosf(0.5f * (q2 - support[s * 4 + 2]))
                * cosf(0.5f * (q3 - support[s * 4 + 3]));
        logit = fmaf(fabsf(p), proj_w[s], logit);
    }
    float pr = 1.0f / (1.0f + expf(-logit));
    out[row * 2 + 0] = pr;
    out[row * 2 + 1] = 1.0f - pr;
}

extern "C" void kernel_launch(void* const* d_in, const int* in_sizes, int n_in,
                              void* d_out, int out_size, void* d_ws, size_t ws_size,
                              hipStream_t stream) {
    (void)in_sizes; (void)n_in; (void)out_size; (void)ws_size;
    const float* x    = (const float*)d_in[0];
    const float* c1w  = (const float*)d_in[1];
    const float* c1b  = (const float*)d_in[2];
    const float* c2w  = (const float*)d_in[3];
    const float* c2b  = (const float*)d_in[4];
    const float* f1w  = (const float*)d_in[5];
    const float* f1b  = (const float*)d_in[6];
    const float* f2w  = (const float*)d_in[7];
    const float* f2b  = (const float*)d_in[8];
    const float* sup  = (const float*)d_in[9];
    const float* pw   = (const float*)d_in[10];
    const float* pb   = (const float*)d_in[11];
    float* out = (float*)d_out;

    // Workspace layout (bytes, all 16B aligned):
    //   pool1 : 128*6*123*123*4  = 46,476,288
    //   A16   : 128*57344*2      = 14,680,064
    //   W16   : 128*57344*2      = 14,680,064
    //   Cpart : 64*128*128*4     =  4,194,304
    //   Cfin  : 128*128*4        =     65,536      total ~80.1 MB
    char* ws = (char*)d_ws;
    float*    pool1 = (float*)ws;
    _Float16* a16   = (_Float16*)(ws + 46476288);
    _Float16* w16   = (_Float16*)(ws + 46476288 + 14680064);
    float*    cpart = (float*)(ws + 46476288 + 14680064 + 14680064);
    float*    cfin  = (float*)(ws + 46476288 + 14680064 + 14680064 + 4194304);

    const int n1 = BATCH * 6 * 123 * 123;        // 11,619,072
    const int n2 = BATCH * 15 * 61 * 61;         //  7,144,320
    const long long nw = (long long)NPAD * KPAD; //  7,340,032
    const int npad = BATCH * (KPAD - FC1_K);     //    195,712

    conv1_pool_relu<<<(n1 + 255) / 256, 256, 0, stream>>>(x, c1w, c1b, pool1);
    conv2_pool_relu<<<(n2 + 255) / 256, 256, 0, stream>>>(pool1, c2w, c2b, a16);
    pad_a16<<<(npad + 255) / 256, 256, 0, stream>>>(a16);
    prep_w16<<<(int)((nw + 255) / 256), 256, 0, stream>>>(f1w, w16);
    fc1_wmma<<<8 * 8 * KSPLIT, 32, 0, stream>>>(a16, w16, cpart);
    fc1_reduce<<<(BATCH * NPAD + 255) / 256, 256, 0, stream>>>(cpart, cfin);
    head_kernel<<<1, BATCH, 0, stream>>>(cfin, f1b, f2w, f2b, sup, pw, pb, out);
}
